// Decoder_42288247996651
// MI455X (gfx1250) — matrix operand order repaired and checked by reading, not compiled, not run on hardware
//
#include <hip/hip_runtime.h>

// ---------------------------------------------------------------------------
// Attention-LSTM decoder for MI455X (gfx1250, wave32, WMMA).
// Strategy: convert all GEMM operands to bf16 once (prologue), run every
// matmul through v_wmma_f32_16x16x32_bf16 (f32 accum), keep all pointwise
// math (softmax / tanh / sigmoid / biases) in fp32.
// Cache policy: embed_bf (16.4MB) + Uv (12.8MB) + weights stay L2-resident
// across all 31 steps; the 254MB logits output is stored non-temporally so
// the write stream does not evict them.
// ---------------------------------------------------------------------------

typedef unsigned short u16;
typedef __attribute__((ext_vector_type(16))) __bf16 v16bf;
typedef __attribute__((ext_vector_type(8)))  float  v8f;

#define BB    64
#define NN    196
#define TT    32
#define TS    31          // T-1 decode steps
#define VOCAB 32000
#define EMB   256
#define HDIM  512
#define VDIM  256
#define ATT   256
#define GATES 2048        // 4*HDIM
#define KPACK 1024        // EMB + VDIM + HDIM

#if defined(__gfx1250__) && \
    __has_builtin(__builtin_amdgcn_global_load_async_to_lds_b32) && \
    __has_builtin(__builtin_amdgcn_s_wait_asynccnt)
#define HAVE_ASYNC_LDS 1
#else
#define HAVE_ASYNC_LDS 0
#endif

union Frag { uint4 u[2]; v16bf v; };

__device__ __forceinline__ u16 f2bf(float f) {
  unsigned int u = __float_as_uint(f);
  u += 0x7FFFu + ((u >> 16) & 1u);       // round-to-nearest-even
  return (u16)(u >> 16);
}

// --------------------------- prologue kernels ------------------------------

// plain f32 -> bf16 convert
__global__ void cv_kernel(const float* __restrict__ src, u16* __restrict__ dst,
                          long count) {
  long i = (long)blockIdx.x * blockDim.x + threadIdx.x;
  if (i < count) dst[i] = f2bf(src[i]);
}

// transpose + convert: src is KxN row-major f32, dst[n*dst_ld + dst_off + k]
__global__ void tc_kernel(const float* __restrict__ src, u16* __restrict__ dst,
                          int K, int N, int dst_ld, int dst_off) {
  long idx = (long)blockIdx.x * blockDim.x + threadIdx.x;
  if (idx >= (long)K * N) return;
  int k = (int)(idx % K);
  int n = (int)(idx / K);
  dst[(long)n * dst_ld + dst_off + k] = f2bf(src[(long)k * N + n]);
}

__global__ void bias_sum_kernel(const float* __restrict__ b_ih,
                                const float* __restrict__ b_hh,
                                float* __restrict__ b_sum) {
  int i = blockIdx.x * blockDim.x + threadIdx.x;
  if (i < GATES) b_sum[i] = b_ih[i] + b_hh[i];
}

// feat_mean = mean_n V[b,n,:]  -> bf16 (A operand of init GEMMs)
__global__ void mean_kernel(const float* __restrict__ V, u16* __restrict__ fm) {
  int idx = blockIdx.x * blockDim.x + threadIdx.x;   // 64*256
  int b = idx >> 8, d = idx & 255;
  float s = 0.f;
  const float* vb = V + (long)b * NN * VDIM + d;
  for (int n = 0; n < NN; ++n) s += vb[(long)n * VDIM];
  fm[idx] = f2bf(s * (1.0f / 196.0f));
}

// --------------------------- generic WMMA GEMM -----------------------------
// C[M x N] = A[M x K](bf16,row) . B[N x K](bf16,row)^T  (+bias) (+tanh)
// grid = (N/64, M/16), block = 128 (4 waves, one 16x16 tile each)
__global__ void __launch_bounds__(128)
gemm_bf16_kernel(const u16* __restrict__ A, const u16* __restrict__ B,
                 const float* __restrict__ bias,
                 float* __restrict__ C, u16* __restrict__ Cbf,
                 int K, int N, long ldc, int act) {
  const int wave = threadIdx.x >> 5;
  const int lane = threadIdx.x & 31;
  const int half = lane >> 4;
  const int r    = lane & 15;
  const int m0   = blockIdx.y << 4;
  const int n0   = (blockIdx.x * 4 + wave) << 4;

  const u16* arow = A + (long)(m0 + r) * K;
  const u16* brow = B + (long)(n0 + r) * K;

  v8f acc = {0.f, 0.f, 0.f, 0.f, 0.f, 0.f, 0.f, 0.f};
  for (int kc = 0; kc < K; kc += 32) {
    Frag a, b;
    a.u[0] = *(const uint4*)(arow + kc + half * 8);        // K: kc..kc+7 | +8..15
    a.u[1] = *(const uint4*)(arow + kc + 16 + half * 8);   // K: +16..23  | +24..31
    b.u[0] = *(const uint4*)(brow + kc + half * 8);        // K: kc..kc+7 | +8..15
    b.u[1] = *(const uint4*)(brow + kc + 16 + half * 8);   // K: +16..23  | +24..31
    acc = __builtin_amdgcn_wmma_f32_16x16x32_bf16(
        false, a.v, false, b.v, (short)0, acc, false, false);
  }

  const float bv = bias ? bias[n0 + r] : 0.0f;
#pragma unroll
  for (int i = 0; i < 8; ++i) {
    int row = m0 + i + half * 8;
    float v = acc[i] + bv;
    if (act == 1) v = tanhf(v);
    if (C)   C[(long)row * ldc + n0 + r] = v;
    if (Cbf) Cbf[(long)row * N + n0 + r] = f2bf(v);
  }
}

// --------------------------- logits GEMM (dominant) ------------------------
// out[b, t, v] slice: C = h_proj(64x256) . embed_bf(32000x256)^T
// A fragments for all 8 K-chunks preloaded; each wave does 4 N tiles.
// Output stored non-temporally (254MB write-once stream must not evict the
// L2-resident embed_bf). grid = (125, 4), block = 128.
__global__ void __launch_bounds__(128)
logits_gemm_kernel(const u16* __restrict__ A, const u16* __restrict__ B,
                   float* __restrict__ C, long ldc) {
  const int wave = threadIdx.x >> 5;
  const int lane = threadIdx.x & 31;
  const int half = lane >> 4;
  const int r    = lane & 15;
  const int m0   = blockIdx.y << 4;

  const u16* arow = A + (long)(m0 + r) * EMB;
  Frag a[8];
#pragma unroll
  for (int kk = 0; kk < 8; ++kk) {
    a[kk].u[0] = *(const uint4*)(arow + kk * 32 + half * 8);
    a[kk].u[1] = *(const uint4*)(arow + kk * 32 + 16 + half * 8);
  }

#pragma unroll
  for (int j = 0; j < 4; ++j) {
    const int n0 = (blockIdx.x * 16 + wave * 4 + j) << 4;
    const u16* brow = B + (long)(n0 + r) * EMB + half * 8;
    if (j < 3)  // pull next tile's embed rows toward L0/L2
      __builtin_prefetch(B + (long)(n0 + 16 + r) * EMB, 0, 1);
    v8f acc = {0.f, 0.f, 0.f, 0.f, 0.f, 0.f, 0.f, 0.f};
#pragma unroll
    for (int kk = 0; kk < 8; ++kk) {
      Frag b;
      b.u[0] = *(const uint4*)(brow + kk * 32);
      b.u[1] = *(const uint4*)(brow + kk * 32 + 16);
      acc = __builtin_amdgcn_wmma_f32_16x16x32_bf16(
          false, a[kk].v, false, b.v, (short)0, acc, false, false);
    }
#pragma unroll
    for (int i = 0; i < 8; ++i)
      __builtin_nontemporal_store(
          acc[i], &C[(long)(m0 + i + half * 8) * ldc + n0 + r]);
  }
}

// --------------------------- per-step pointwise kernels --------------------

// attention: e = v_att . tanh(hW[b] + Uv[b,n]); softmax over n; ctx = a.V
// grid = 64, block = 256. LDS staging of hW/v_att goes through the gfx1250
// async-to-LDS path (ASYNCcnt) when the toolchain exposes it.
__global__ void attn_kernel(const float* __restrict__ hW,
                            const float* __restrict__ Uv,
                            const float* __restrict__ v_att,
                            const float* __restrict__ V,
                            float* __restrict__ ctx) {
  const int b = blockIdx.x, tid = threadIdx.x;
  __shared__ float s_h[ATT], s_v[ATT], red[256], s_a[NN];
#if HAVE_ASYNC_LDS
  __builtin_amdgcn_global_load_async_to_lds_b32(
      (__attribute__((address_space(1))) int*)(hW + b * ATT + tid),
      (__attribute__((address_space(3))) int*)&s_h[tid], 0, 0);
  __builtin_amdgcn_global_load_async_to_lds_b32(
      (__attribute__((address_space(1))) int*)(v_att + tid),
      (__attribute__((address_space(3))) int*)&s_v[tid], 0, 0);
  __builtin_amdgcn_s_wait_asynccnt(0);
  __syncthreads();
#else
  s_h[tid] = hW[b * ATT + tid];
  s_v[tid] = v_att[tid];
  __syncthreads();
#endif

  float e = -1e30f;
  if (tid < NN) {
    const float* uv = Uv + ((long)b * NN + tid) * ATT;
    float acc = 0.f;
    for (int a = 0; a < ATT; ++a) acc += s_v[a] * tanhf(s_h[a] + uv[a]);
    e = acc;
  }
  red[tid] = e; __syncthreads();
  for (int s = 128; s > 0; s >>= 1) {
    if (tid < s) red[tid] = fmaxf(red[tid], red[tid + s]);
    __syncthreads();
  }
  const float m = red[0]; __syncthreads();
  const float ex = (tid < NN) ? __expf(e - m) : 0.f;
  red[tid] = ex; __syncthreads();
  for (int s = 128; s > 0; s >>= 1) {
    if (tid < s) red[tid] += red[tid + s];
    __syncthreads();
  }
  const float inv = 1.f / red[0];
  if (tid < NN) s_a[tid] = ex * inv;
  __syncthreads();

  float acc = 0.f;                       // tid = feature dim d (coalesced per n)
  const float* vb = V + (long)b * NN * VDIM + tid;
  for (int n = 0; n < NN; ++n) acc += s_a[n] * vb[(long)n * VDIM];
  ctx[b * VDIM + tid] = acc;
}

// A_pack[b, :] = bf16([ embed[y[b,t]] | ctx[b] | h[b] ]); grid = 256x256
__global__ void pack_kernel(const float* __restrict__ embed,
                            const int* __restrict__ y, int t,
                            const float* __restrict__ ctx,
                            const float* __restrict__ h,
                            u16* __restrict__ Apack) {
  int idx = blockIdx.x * blockDim.x + threadIdx.x;   // 64*1024
  int b = idx >> 10, k = idx & 1023;
  float v;
  if (k < EMB)             v = embed[(long)y[b * TT + t] * EMB + k];
  else if (k < EMB + VDIM) v = ctx[b * VDIM + (k - EMB)];
  else                     v = h[b * HDIM + (k - EMB - VDIM)];
  Apack[idx] = f2bf(v);
}

// LSTM cell update (gates already include b_ih+b_hh); grid = 128x256
__global__ void lstm_kernel(const float* __restrict__ gates,
                            float* __restrict__ h, float* __restrict__ c,
                            u16* __restrict__ h_bf) {
  int idx = blockIdx.x * blockDim.x + threadIdx.x;   // 64*512
  int b = idx >> 9, j = idx & 511;
  const float* g = gates + (long)b * GATES;
  float gi = g[j], gf = g[HDIM + j], gg = g[2 * HDIM + j], go = g[3 * HDIM + j];
  float si = 1.f / (1.f + __expf(-gi));
  float sf = 1.f / (1.f + __expf(-gf));
  float so = 1.f / (1.f + __expf(-go));
  float cn = sf * c[idx] + si * tanhf(gg);
  float hn = so * tanhf(cn);
  c[idx] = cn;
  h[idx] = hn;
  h_bf[idx] = f2bf(hn);
}

// --------------------------- host-side launch ------------------------------

extern "C" void kernel_launch(void* const* d_in, const int* in_sizes, int n_in,
                              void* d_out, int out_size, void* d_ws, size_t ws_size,
                              hipStream_t stream) {
  const float* V       = (const float*)d_in[0];
  const int*   y       = (const int*)  d_in[1];
  const float* embed   = (const float*)d_in[2];
  const float* W_att   = (const float*)d_in[3];
  const float* U_att   = (const float*)d_in[4];
  const float* v_att   = (const float*)d_in[5];
  const float* W_ih    = (const float*)d_in[6];
  const float* W_hh    = (const float*)d_in[7];
  const float* b_ih    = (const float*)d_in[8];
  const float* b_hh    = (const float*)d_in[9];
  const float* W_inith = (const float*)d_in[10];
  const float* b_inith = (const float*)d_in[11];
  const float* W_initc = (const float*)d_in[12];
  const float* b_initc = (const float*)d_in[13];
  const float* W_proj  = (const float*)d_in[14];
  float* out = (float*)d_out;

  // workspace carve (~42 MB)
  char* p = (char*)d_ws;
  auto alloc = [&](size_t bytes) {
    void* r = (void*)p;
    p += (bytes + 255) & ~(size_t)255;
    return r;
  };
  u16*   embed_bf = (u16*)  alloc((size_t)VOCAB * EMB * 2);
  u16*   V_bf     = (u16*)  alloc((size_t)BB * NN * VDIM * 2);
  u16*   Wt_all   = (u16*)  alloc((size_t)GATES * KPACK * 2);  // [j,0:512]=W_ih^T, [512:1024]=W_hh^T
  u16*   Watt_t   = (u16*)  alloc((size_t)ATT * HDIM * 2);
  u16*   Uatt_t   = (u16*)  alloc((size_t)ATT * VDIM * 2);
  u16*   Wh0_t    = (u16*)  alloc((size_t)HDIM * VDIM * 2);
  u16*   Wc0_t    = (u16*)  alloc((size_t)HDIM * VDIM * 2);
  u16*   Wp_t     = (u16*)  alloc((size_t)EMB * HDIM * 2);
  float* b_sum    = (float*)alloc((size_t)GATES * 4);
  u16*   fm_bf    = (u16*)  alloc((size_t)BB * VDIM * 2);
  float* Uv       = (float*)alloc((size_t)BB * NN * ATT * 4);
  float* h        = (float*)alloc((size_t)BB * HDIM * 4);
  float* c        = (float*)alloc((size_t)BB * HDIM * 4);
  u16*   h_bf     = (u16*)  alloc((size_t)BB * HDIM * 2);
  float* hW       = (float*)alloc((size_t)BB * ATT * 4);
  float* ctx      = (float*)alloc((size_t)BB * VDIM * 4);
  u16*   Apack    = (u16*)  alloc((size_t)BB * KPACK * 2);
  float* gates    = (float*)alloc((size_t)BB * GATES * 4);
  u16*   hproj_bf = (u16*)  alloc((size_t)BB * EMB * 2);
  (void)ws_size; (void)in_sizes; (void)n_in; (void)out_size;

  auto gemm = [&](const u16* A, const u16* Bm, const float* bias, float* C,
                  u16* Cbf, int M, int K, int N, long ldc, int act) {
    gemm_bf16_kernel<<<dim3(N / 64, M / 16), 128, 0, stream>>>(
        A, Bm, bias, C, Cbf, K, N, ldc, act);
  };

  // ---------- prologue: converts / transposes / precompute ----------
  {
    long n1 = (long)VOCAB * EMB;
    cv_kernel<<<(unsigned)((n1 + 255) / 256), 256, 0, stream>>>(embed, embed_bf, n1);
    long n2 = (long)BB * NN * VDIM;
    cv_kernel<<<(unsigned)((n2 + 255) / 256), 256, 0, stream>>>(V, V_bf, n2);
  }
  auto tc = [&](const float* src, u16* dst, int K, int N, int ld, int off) {
    long cnt = (long)K * N;
    tc_kernel<<<(unsigned)((cnt + 255) / 256), 256, 0, stream>>>(src, dst, K, N, ld, off);
  };
  tc(W_ih,    Wt_all, 512, GATES, KPACK, 0);
  tc(W_hh,    Wt_all, 512, GATES, KPACK, 512);
  tc(W_att,   Watt_t, HDIM, ATT, HDIM, 0);
  tc(U_att,   Uatt_t, VDIM, ATT, VDIM, 0);
  tc(W_inith, Wh0_t,  VDIM, HDIM, VDIM, 0);
  tc(W_initc, Wc0_t,  VDIM, HDIM, VDIM, 0);
  tc(W_proj,  Wp_t,   HDIM, EMB, HDIM, 0);
  bias_sum_kernel<<<GATES / 256, 256, 0, stream>>>(b_ih, b_hh, b_sum);
  mean_kernel<<<(BB * VDIM) / 256, 256, 0, stream>>>(V, fm_bf);

  // Uv = V . U_att   (12544 x 256, K=256)
  gemm(V_bf, Uatt_t, nullptr, Uv, nullptr, BB * NN, VDIM, ATT, ATT, 0);
  // h0 = tanh(fm @ W_inith + b),  c0 = tanh(fm @ W_initc + b)
  gemm(fm_bf, Wh0_t, b_inith, h, h_bf, BB, VDIM, HDIM, HDIM, 1);
  gemm(fm_bf, Wc0_t, b_initc, c, nullptr, BB, VDIM, HDIM, HDIM, 1);

  // ---------- 31 decode steps ----------
  for (int t = 0; t < TS; ++t) {
    // hW = h @ W_att          (64 x 256, K=512)
    gemm(h_bf, Watt_t, nullptr, hW, nullptr, BB, HDIM, ATT, ATT, 0);
    // attention -> ctx
    attn_kernel<<<BB, 256, 0, stream>>>(hW, Uv, v_att, V, ctx);
    // A_pack = bf16([x_t | ctx | h])
    pack_kernel<<<(BB * KPACK) / 256, 256, 0, stream>>>(embed, y, t, ctx, h, Apack);
    // gates = A_pack @ [W_ih;W_hh] + (b_ih + b_hh)   (64 x 2048, K=1024)
    gemm(Apack, Wt_all, b_sum, gates, nullptr, BB, KPACK, GATES, GATES, 0);
    // LSTM pointwise -> h, c, h_bf
    lstm_kernel<<<(BB * HDIM) / 256, 256, 0, stream>>>(gates, h, c, h_bf);
    // h_proj = h @ W_proj -> bf16   (64 x 256, K=512)
    gemm(h_bf, Wp_t, nullptr, nullptr, hproj_bf, BB, HDIM, EMB, EMB, 0);
    // logits_t = h_proj @ embed^T  -> out[:, t, :]  (64 x 32000, K=256)
    logits_gemm_kernel<<<dim3(VOCAB / 256, BB / 16), 128, 0, stream>>>(
        hproj_bf, embed_bf, out + (long)t * VOCAB, (long)TS * VOCAB);
  }
}
